// RPN_39951785787757
// MI455X (gfx1250) — matrix operand order repaired and checked by reading, not compiled, not run on hardware
//
#include <hip/hip_runtime.h>
#include <hip/hip_bf16.h>
#include <stdint.h>

// ---------------------------------------------------------------------------
// RPN forward for MI455X (gfx1250, wave32, WMMA + async-to-LDS staging).
//   inputs : features[8,256,64,64] f32, conv_w[512,256,3,3], conv_b[512],
//            cls_w[18,512,1,1], cls_b[18], reg_w[36,512,1,1], reg_b[36],
//            anchors[36864,4], img_size (int scalar)
//   outputs (concat f32): cls_out[8,36864,2], reg_out[8,36864,4],
//                         proposals[8,1000,4]
// ---------------------------------------------------------------------------

typedef __attribute__((ext_vector_type(8)))  _Float16 v8h;
typedef __attribute__((ext_vector_type(16))) _Float16 v16h;
typedef __attribute__((ext_vector_type(8)))  float    v8f;

#define B_      8
#define CIN_    256
#define CMID_   512
#define HW_     64
#define NANCH_  36864      // 64*64*9
#define PRE_    2000
#define POST_   1000
#define SORTN_  65536

// d_out float offsets
#define OUT_CLS_  0
#define OUT_REG_  (B_*NANCH_*2)                 // 589824
#define OUT_PROP_ (OUT_REG_ + B_*NANCH_*4)     // 1769472

// workspace byte offsets (all 256-aligned)
#define WS_XPAD   0ull                                  // f16 [8][66][66][256]
#define WS_WT     17842176ull                           // f16 [10][512][256] (tap 9 = prefetch pad)
#define WS_HWT    (WS_WT + 2621440ull)                  // f16 [64][512]
#define WS_HB     (WS_HWT + 65536ull)                   // f32 [64]
#define WS_X      (WS_HB + 256ull)                      // f16 [8*4096][512]
#define WS_BOXES  (WS_X + 33554432ull)                  // f32 [8][36864][4]
#define WS_KEYS   (WS_BOXES + 4718592ull)               // u64 [8][65536]
#define WS_TOPBOX (WS_KEYS + 4194304ull)                // f32 [8][2000][4]
#define WS_TOPVAL (WS_TOPBOX + 256000ull)               // u32 [8][2000]
#define WS_SUP    (WS_TOPVAL + 64000ull)                // u32 [8][2000][64]

static __device__ inline v16h cat16(v8h a, v8h b) {
  return __builtin_shufflevector(a, b, 0,1,2,3,4,5,6,7,8,9,10,11,12,13,14,15);
}

// CDNA5 async global->LDS copy (16B per lane) + completion wait (ASYNCcnt)
static __device__ inline void async_cp16(void* lds, const void* g) {
  unsigned l = (unsigned)(uintptr_t)lds;                 // LDS byte offset
  unsigned long long a = (unsigned long long)(uintptr_t)g;
  asm volatile("global_load_async_to_lds_b128 %0, %1, off"
               :: "v"(l), "v"(a) : "memory");
}
static __device__ inline void wait_async0() {
  asm volatile("s_wait_asynccnt 0x0" ::: "memory");
}

// -------------------- feature f32 NCHW -> f16 NHWC (+1 halo) ----------------
__global__ void k_cvt_feat(const float* __restrict__ f, _Float16* __restrict__ xpad) {
  unsigned idx = blockIdx.x * blockDim.x + threadIdx.x;   // < 8*66*66*32
  unsigned c0 = (idx & 31u) * 8u;
  unsigned wp = (idx >> 5) % 66u;
  unsigned hp = (idx / (32u * 66u)) % 66u;
  unsigned b  = idx / (32u * 66u * 66u);
  v8h v;
  if (hp >= 1 && hp <= 64 && wp >= 1 && wp <= 64) {
    unsigned h = hp - 1, w = wp - 1;
    #pragma unroll
    for (int k = 0; k < 8; ++k)
      v[k] = (_Float16)f[(((size_t)b * CIN_ + c0 + k) * HW_ + h) * HW_ + w];
  } else {
    #pragma unroll
    for (int k = 0; k < 8; ++k) v[k] = (_Float16)0.f;
  }
  *(v8h*)(xpad + (((size_t)b * 66 + hp) * 66 + wp) * CIN_ + c0) = v;
}

// -------------------- conv weights -> f16 [tap][cout][cin] ------------------
__global__ void k_cvt_w(const float* __restrict__ w, _Float16* __restrict__ wt) {
  unsigned idx = blockIdx.x * blockDim.x + threadIdx.x;   // < 9*512*32
  unsigned c0   = (idx & 31u) * 8u;
  unsigned cout = (idx >> 5) & 511u;
  unsigned tap  = idx >> 14;
  v8h v;
  #pragma unroll
  for (int k = 0; k < 8; ++k)
    v[k] = (_Float16)w[((size_t)cout * CIN_ + c0 + k) * 9 + tap];
  *(v8h*)(wt + ((size_t)tap * CMID_ + cout) * CIN_ + c0) = v;
}

// -------------------- head weights (18 cls + 36 reg -> 64 pad) --------------
__global__ void k_cvt_head(const float* __restrict__ cw, const float* __restrict__ cb,
                           const float* __restrict__ rw, const float* __restrict__ rb,
                           _Float16* __restrict__ hwt, float* __restrict__ hb) {
  unsigned idx = blockIdx.x * blockDim.x + threadIdx.x;   // < 64*64
  unsigned c0 = (idx & 63u) * 8u;
  unsigned co = idx >> 6;
  v8h v;
  #pragma unroll
  for (int k = 0; k < 8; ++k) {
    float x = 0.f;
    if (co < 18)      x = cw[(size_t)co * CMID_ + c0 + k];
    else if (co < 54) x = rw[(size_t)(co - 18) * CMID_ + c0 + k];
    v[k] = (_Float16)x;
  }
  *(v8h*)(hwt + (size_t)co * CMID_ + c0) = v;
  if (c0 == 0) hb[co] = (co < 18) ? cb[co] : (co < 54 ? rb[co - 18] : 0.f);
}

// -------------------- 3x3 conv + bias + ReLU via WMMA -----------------------
// block = 256 thr (8 waves); covers 16 pixels (one w-strip) x 512 couts.
// Activation halo patch staged in LDS via async copy; weight fragments
// software-pipelined with two statically-named buffers (no dynamic indexing
// => no scratch spill). WT has a 10th dummy tap so ks+2 prefetch is
// unconditional (the ks=70 prefetch is never consumed).
__global__ void __launch_bounds__(256)
k_conv3(const _Float16* __restrict__ xpad, const _Float16* __restrict__ wt,
        const float* __restrict__ bias, _Float16* __restrict__ xo) {
  __shared__ _Float16 patch[3 * 18 * 256];        // 27.6 KB
  unsigned blk = blockIdx.x;                       // < 8*64*4
  unsigned b  = blk >> 8;
  unsigned h  = (blk >> 2) & 63u;
  unsigned w0 = (blk & 3u) * 16u;
  unsigned tid = threadIdx.x;
  unsigned wave = tid >> 5;
  unsigned lane = tid & 31u;
  unsigned m   = lane & 15u;
  unsigned sel = lane >> 4;
  unsigned cout0 = wave * 64u;

  // stage 3 rows x (18 cols * 256 cin) f16 -> LDS (rows contiguous in global)
  #pragma unroll
  for (int r = 0; r < 3; ++r) {
    const _Float16* src = xpad + (((size_t)b * 66 + (h + r)) * 66 + w0) * CIN_;
    _Float16* dst = patch + r * 18 * 256;
    for (unsigned j = tid; j < 576; j += 256)      // 576 x 16B = 9216 B / row
      async_cp16(dst + j * 8, src + j * 8);
  }
  wait_async0();
  __syncthreads();

  v8f acc[4];
  #pragma unroll
  for (int t = 0; t < 4; ++t) acc[t] = (v8f)0.f;

  const _Float16* wbase = wt + ((size_t)cout0 + (lane & 15u)) * CIN_ + sel * 16;
  // B fragment address for k-step ks (tap = ks>>3, c0 = (ks&7)*32), tile t
  #define BPTR(ks, t) (wbase + ((size_t)((unsigned)(ks) >> 3) * CMID_ + (t) * 16) * CIN_ \
                             + (((unsigned)(ks) & 7u) << 5))

  v16h B0[4], B1[4];
  #pragma unroll
  for (int t = 0; t < 4; ++t) B0[t] = *(const v16h*)BPTR(0, t);

  for (int ks = 0; ks < 72; ks += 2) {
    int tap = ks >> 3, c0 = (ks & 7) << 5;
    int dy = tap / 3, dx = tap % 3;
    const _Float16* aprow = patch + dy * (18 * 256) + (dx + m) * 256;

    // prefetch odd step's weights
    #pragma unroll
    for (int t = 0; t < 4; ++t) B1[t] = *(const v16h*)BPTR(ks + 1, t);

    // even step
    {
      const _Float16* ap = aprow + c0;
      v16h af = cat16(*(const v8h*)(ap + sel * 16), *(const v8h*)(ap + sel * 16 + 8));
      #pragma unroll
      for (int t = 0; t < 4; ++t)
        acc[t] = __builtin_amdgcn_wmma_f32_16x16x32_f16(
            false, af, false, B0[t], (short)0, acc[t], false, false);
    }

    // prefetch next pair's even weights (ks=70 reads the dummy tap; unused)
    #pragma unroll
    for (int t = 0; t < 4; ++t) B0[t] = *(const v16h*)BPTR(ks + 2, t);

    // odd step
    {
      const _Float16* ap = aprow + c0 + 32;
      v16h af = cat16(*(const v8h*)(ap + sel * 16), *(const v8h*)(ap + sel * 16 + 8));
      #pragma unroll
      for (int t = 0; t < 4; ++t)
        acc[t] = __builtin_amdgcn_wmma_f32_16x16x32_f16(
            false, af, false, B1[t], (short)0, acc[t], false, false);
    }
  }
  #undef BPTR

  // epilogue: bias + relu -> f16 NHWC
  #pragma unroll
  for (int t = 0; t < 4; ++t) {
    unsigned co = cout0 + t * 16 + (lane & 15u);
    float bb = bias[co];
    #pragma unroll
    for (int r = 0; r < 8; ++r) {
      unsigned px = (sel ? 8u : 0u) + r;
      float v = acc[t][r] + bb;
      v = v > 0.f ? v : 0.f;
      xo[(((size_t)b * HW_ + h) * HW_ + (w0 + px)) * CMID_ + co] = (_Float16)v;
    }
  }
}

// -------------------- fused 1x1 heads via WMMA ------------------------------
// block = 256 thr (8 waves) covers 2 pixel-tiles x 4 cout-tiles (64 couts).
// x-tiles staged in LDS (async); K loop fully unrolled (constant offsets) so
// the compiler hoists/clusters the weight loads itself.
__global__ void __launch_bounds__(256)
k_head(const _Float16* __restrict__ x, const _Float16* __restrict__ hwt,
       const float* __restrict__ hb, float* __restrict__ out) {
  __shared__ _Float16 xt[2][16 * 512];            // 32 KB
  unsigned tid = threadIdx.x;
  unsigned wave = tid >> 5;
  unsigned lane = tid & 31u;
  unsigned pt = wave >> 2, t = wave & 3u;
  unsigned tile0 = blockIdx.x * 2;                 // pixel tiles (of 2048)
  unsigned m = lane & 15u, sel = lane >> 4;

  // both tiles are contiguous in x: copy 32 KB
  const _Float16* src = x + (size_t)tile0 * 16 * CMID_;
  for (unsigned j = tid; j < 2048; j += 256)       // 2048 x 16B
    async_cp16(&xt[0][0] + j * 8, src + j * 8);
  wait_async0();
  __syncthreads();

  const _Float16* wrow = hwt + (size_t)(t * 16 + (lane & 15u)) * CMID_ + sel * 16;
  const _Float16* arow = &xt[pt][0] + m * CMID_;
  v8f acc = (v8f)0.f;
  #pragma unroll
  for (int ks = 0; ks < 16; ++ks) {
    v16h bf = *(const v16h*)(wrow + (ks << 5));
    const _Float16* ap = arow + (ks << 5);
    v16h af = cat16(*(const v8h*)(ap + sel * 16), *(const v8h*)(ap + sel * 16 + 8));
    acc = __builtin_amdgcn_wmma_f32_16x16x32_f16(
        false, af, false, bf, (short)0, acc, false, false);
  }

  unsigned tile = tile0 + pt;
  unsigned co = t * 16 + (lane & 15u);
  float bb = hb[co];
  #pragma unroll
  for (int r = 0; r < 8; ++r) {
    unsigned p = tile * 16 + (sel ? 8u : 0u) + r;
    unsigned b = p >> 12, hw = p & 4095u;
    float v = acc[r] + bb;
    if (co < 18) {
      unsigned a = co >> 1, c = co & 1u;
      out[OUT_CLS_ + ((size_t)b * NANCH_ + hw * 9 + a) * 2 + c] = v;
    } else if (co < 54) {
      unsigned cc = co - 18, a = cc >> 2, rc = cc & 3u;
      out[OUT_REG_ + ((size_t)b * NANCH_ + hw * 9 + a) * 4 + rc] = v;
    }
  }
}

// -------------------- score / box decode / sort keys ------------------------
__global__ void k_decode(const float* __restrict__ out, const float* __restrict__ anchors,
                         const int* __restrict__ imgsz, float* __restrict__ boxes,
                         unsigned long long* __restrict__ keys) {
  unsigned idx = blockIdx.x * blockDim.x + threadIdx.x;   // < 8*65536
  unsigned b = idx >> 16, i = idx & 65535u;
  if (i >= NANCH_) { keys[((size_t)b << 16) + i] = 0ull; return; }
  float img = (float)imgsz[0];
  float s0 = out[OUT_CLS_ + ((size_t)b * NANCH_ + i) * 2 + 0];
  float s1 = out[OUT_CLS_ + ((size_t)b * NANCH_ + i) * 2 + 1];
  float score = 1.f / (1.f + __expf(s0 - s1));
  const float* rg = out + OUT_REG_ + ((size_t)b * NANCH_ + i) * 4;
  float a0 = anchors[i * 4 + 0], a1 = anchors[i * 4 + 1];
  float a2 = anchors[i * 4 + 2], a3 = anchors[i * 4 + 3];
  float wa = a2 - a0, ha = a3 - a1;
  float cx = rg[0] * wa + a0 + 0.5f * wa;
  float cy = rg[1] * ha + a1 + 0.5f * ha;
  float ww = __expf(rg[2]) * wa, hh = __expf(rg[3]) * ha;
  float x1 = fminf(fmaxf(cx - 0.5f * ww, 0.f), img);
  float y1 = fminf(fmaxf(cy - 0.5f * hh, 0.f), img);
  float x2 = fminf(fmaxf(cx + 0.5f * ww, 0.f), img);
  float y2 = fminf(fmaxf(cy + 0.5f * hh, 0.f), img);
  float* bx = boxes + ((size_t)b * NANCH_ + i) * 4;
  bx[0] = x1; bx[1] = y1; bx[2] = x2; bx[3] = y2;
  bool valid = (x2 - x1 >= 16.f) && (y2 - y1 >= 16.f);
  float ms = valid ? score : -__builtin_huge_valf();
  unsigned u = __float_as_uint(ms);
  unsigned mono = (u & 0x80000000u) ? ~u : (u | 0x80000000u);
  keys[((size_t)b << 16) + i] = ((unsigned long long)mono << 32) | (unsigned)(~i);
}

// -------------------- global bitonic sort (desc) ----------------------------
__global__ void k_bitonic(unsigned long long* __restrict__ keys, unsigned k, unsigned j) {
  unsigned gid = blockIdx.x * blockDim.x + threadIdx.x;   // < 8*65536
  unsigned b = gid >> 16, i = gid & 65535u;
  unsigned ixj = i ^ j;
  if (ixj <= i) return;
  unsigned long long* kb = keys + ((size_t)b << 16);
  unsigned long long a = kb[i], c = kb[ixj];
  bool desc = ((i & k) == 0);
  if ((a < c) == desc) { kb[i] = c; kb[ixj] = a; }
}

// -------------------- gather top-2000 boxes ---------------------------------
__global__ void k_gather(const unsigned long long* __restrict__ keys,
                         const float* __restrict__ boxes,
                         float* __restrict__ topbox, unsigned* __restrict__ topval) {
  unsigned idx = blockIdx.x * blockDim.x + threadIdx.x;
  if (idx >= B_ * PRE_) return;
  unsigned b = idx / PRE_, kk = idx % PRE_;
  unsigned long long key = keys[((size_t)b << 16) + kk];
  unsigned mono = (unsigned)(key >> 32);
  unsigned i = ~(unsigned)(key & 0xFFFFFFFFull);
  bool valid = (mono > 0x007FFFFFu) && (i < NANCH_);
  float* tb = topbox + ((size_t)b * PRE_ + kk) * 4;
  if (valid) {
    const float* bx = boxes + ((size_t)b * NANCH_ + i) * 4;
    tb[0] = bx[0]; tb[1] = bx[1]; tb[2] = bx[2]; tb[3] = bx[3];
  } else {
    tb[0] = tb[1] = tb[2] = tb[3] = 0.f;
  }
  topval[(size_t)b * PRE_ + kk] = valid ? 1u : 0u;
}

// -------------------- IoU suppression bitmask -------------------------------
__global__ void k_ioumask(const float* __restrict__ topbox, unsigned* __restrict__ sup) {
  unsigned idx = blockIdx.x * blockDim.x + threadIdx.x;   // < 8*2000*63
  if (idx >= B_ * PRE_ * 63) return;
  unsigned wch = idx % 63u;
  unsigned i = (idx / 63u) % PRE_;
  unsigned b = idx / (63u * PRE_);
  const float* bi = topbox + ((size_t)b * PRE_ + i) * 4;
  float ax1 = bi[0], ay1 = bi[1], ax2 = bi[2], ay2 = bi[3];
  float areaA = (ax2 - ax1) * (ay2 - ay1);
  unsigned bits = 0u;
  unsigned j0 = wch * 32u;
  #pragma unroll 4
  for (unsigned jj = 0; jj < 32; ++jj) {
    unsigned j = j0 + jj;
    if (j >= PRE_ || j <= i) continue;
    const float* bj = topbox + ((size_t)b * PRE_ + j) * 4;
    float bx1 = bj[0], by1 = bj[1], bx2 = bj[2], by2 = bj[3];
    float iw = fminf(ax2, bx2) - fmaxf(ax1, bx1);
    float ih = fminf(ay2, by2) - fmaxf(ay1, by1);
    iw = iw > 0.f ? iw : 0.f; ih = ih > 0.f ? ih : 0.f;
    float inter = iw * ih;
    float areaB = (bx2 - bx1) * (by2 - by1);
    float iou = inter / (areaA + areaB - inter + 1e-9f);
    if (iou > 0.7f) bits |= (1u << jj);
  }
  sup[((size_t)b * PRE_ + i) * 64 + wch] = bits;
}

// -------------------- sequential greedy-NMS scan + compaction ---------------
__global__ void __launch_bounds__(64)
k_nms(const unsigned* __restrict__ supmask, const unsigned* __restrict__ topval,
      const float* __restrict__ topbox, float* __restrict__ out) {
  __shared__ unsigned sup[64];
  __shared__ int s_keep, s_slot, s_kc;
  unsigned b = blockIdx.x, tid = threadIdx.x;
  if (tid < 64) sup[tid] = 0u;
  if (tid == 0) s_kc = 0;
  __syncthreads();
  float* prop = out + OUT_PROP_ + (size_t)b * POST_ * 4;
  for (int i = 0; i < PRE_; ++i) {
    if (tid == 0) {
      bool keep = topval[(size_t)b * PRE_ + i] &&
                  !((sup[i >> 5] >> (i & 31)) & 1u);
      s_keep = keep ? 1 : 0;
      s_slot = -1;
      if (keep && s_kc < POST_) { s_slot = s_kc; s_kc++; }
    }
    __syncthreads();
    if (s_keep && tid < 63)
      sup[tid] |= supmask[((size_t)b * PRE_ + i) * 64 + tid];
    if (s_slot >= 0 && tid < 4)
      prop[(size_t)s_slot * 4 + tid] = topbox[((size_t)b * PRE_ + i) * 4 + tid];
    __syncthreads();
  }
  int kc = s_kc;
  for (int r = kc * 4 + tid; r < POST_ * 4; r += 64) prop[r] = 0.f;
}

// ---------------------------------------------------------------------------
extern "C" void kernel_launch(void* const* d_in, const int* in_sizes, int n_in,
                              void* d_out, int out_size, void* d_ws, size_t ws_size,
                              hipStream_t stream) {
  const float* features = (const float*)d_in[0];
  const float* conv_w   = (const float*)d_in[1];
  const float* conv_b   = (const float*)d_in[2];
  const float* cls_w    = (const float*)d_in[3];
  const float* cls_b    = (const float*)d_in[4];
  const float* reg_w    = (const float*)d_in[5];
  const float* reg_b    = (const float*)d_in[6];
  const float* anchors  = (const float*)d_in[7];
  const int*   img_size = (const int*)d_in[8];
  float* out = (float*)d_out;
  char* ws = (char*)d_ws;

  _Float16* xpad = (_Float16*)(ws + WS_XPAD);
  _Float16* wt   = (_Float16*)(ws + WS_WT);
  _Float16* hwt  = (_Float16*)(ws + WS_HWT);
  float*    hb   = (float*)(ws + WS_HB);
  _Float16* x    = (_Float16*)(ws + WS_X);
  float*    boxes= (float*)(ws + WS_BOXES);
  unsigned long long* keys = (unsigned long long*)(ws + WS_KEYS);
  float*    topbox = (float*)(ws + WS_TOPBOX);
  unsigned* topval = (unsigned*)(ws + WS_TOPVAL);
  unsigned* sup    = (unsigned*)(ws + WS_SUP);

  // 1) precision / layout conversion
  k_cvt_feat<<<4356, 256, 0, stream>>>(features, xpad);
  k_cvt_w<<<576, 256, 0, stream>>>(conv_w, wt);
  k_cvt_head<<<16, 256, 0, stream>>>(cls_w, cls_b, reg_w, reg_b, hwt, hb);

  // 2) 3x3 conv + ReLU (WMMA implicit GEMM, LDS-staged activations)
  k_conv3<<<2048, 256, 0, stream>>>(xpad, wt, conv_b, x);

  // 3) fused cls/reg 1x1 heads (WMMA, LDS-staged x tiles)
  k_head<<<1024, 256, 0, stream>>>(x, hwt, hb, out);

  // 4) softmax score + box decode + sort keys
  k_decode<<<2048, 256, 0, stream>>>(out, anchors, img_size, boxes, keys);

  // 5) per-batch descending bitonic sort over 65536 keys
  for (unsigned k = 2; k <= SORTN_; k <<= 1)
    for (unsigned j = k >> 1; j > 0; j >>= 1)
      k_bitonic<<<2048, 256, 0, stream>>>(keys, k, j);

  // 6) gather top-2000, IoU bitmask, greedy NMS, compaction to 1000
  k_gather<<<(B_ * PRE_ + 255) / 256, 256, 0, stream>>>(keys, boxes, topbox, topval);
  k_ioumask<<<(B_ * PRE_ * 63 + 255) / 256, 256, 0, stream>>>(topbox, sup);
  k_nms<<<B_, 64, 0, stream>>>(sup, topval, topbox, out);
}
